// Transformer_17901423689817
// MI455X (gfx1250) — matrix operand order repaired and checked by reading, not compiled, not run on hardware
//
#include <hip/hip_runtime.h>

typedef __bf16 bf16;
typedef bf16  bf16x8  __attribute__((ext_vector_type(8)));
typedef bf16  bf16x16 __attribute__((ext_vector_type(16)));
typedef float f32x8   __attribute__((ext_vector_type(8)));
typedef int   int4v   __attribute__((ext_vector_type(4)));

#define T_SEQ 1024
#define B_SZ  8
#define NH    8
#define DKH   64
#define DM    512
#define MROWS (B_SZ * T_SEQ)   // 8192

// ---------------------------------------------------------------------------
// CDNA5 async global->LDS (ASYNCcnt) with graceful fallback
// ---------------------------------------------------------------------------
#if defined(__has_builtin)
#if __has_builtin(__builtin_amdgcn_global_load_async_to_lds_b128)
#define GEMM_ASYNC 1
#endif
#endif
#ifndef GEMM_ASYNC
#define GEMM_ASYNC 0
#endif

#if GEMM_ASYNC
#if __has_builtin(__builtin_amdgcn_s_wait_asynccnt)
#define WAIT_ASYNC(n) __builtin_amdgcn_s_wait_asynccnt(n)
#else
#define WAIT_ASYNC(n) asm volatile("s_wait_asynccnt %0" ::"i"(n))
#endif
#else
#define WAIT_ASYNC(n)
#endif

typedef __attribute__((address_space(1))) int4v glb_int4;
typedef __attribute__((address_space(3))) int4v lds_int4;

__device__ __forceinline__ void async_copy16(const bf16* g, bf16* l) {
#if GEMM_ASYNC
  __builtin_amdgcn_global_load_async_to_lds_b128(
      (glb_int4*)(void*)g, (lds_int4*)(void*)l, 0, 0);
#else
  *(bf16x8*)l = *(const bf16x8*)g;
#endif
}

// ---------------------------------------------------------------------------
// WMMA helpers (wave32, v_wmma_f32_16x16x32_bf16)
// A 16x32 frag, lane l(<16): row=l, K={0..7,16..23}; lane>=16: row=l-16,
//   K={8..15,24..31}  -> two contiguous 16B chunks per lane.
// B 32x16 frag, lane l: N=l&15, K same as A ({0..7,16..23} / {8..15,24..31}).
// C/D 16x16 f32: vgpr j, lane l -> (M = j + 8*(l>>4), N = l&15).
// ---------------------------------------------------------------------------
__device__ __forceinline__ bf16x16 load_frag(const bf16* p0, const bf16* p1) {
  bf16x8 lo = *(const bf16x8*)p0;
  bf16x8 hi = *(const bf16x8*)p1;
  bf16x16 r;
#pragma unroll
  for (int i = 0; i < 8; ++i) { r[i] = lo[i]; r[i + 8] = hi[i]; }
  return r;
}

__device__ __forceinline__ f32x8 wmma_bf16(bf16x16 a, bf16x16 b, f32x8 c) {
  return __builtin_amdgcn_wmma_f32_16x16x32_bf16(false, a, false, b,
                                                 (short)0, c, false, false);
}

// ---------------------------------------------------------------------------
// Prep kernels
// ---------------------------------------------------------------------------
__global__ void pe_kernel(float* __restrict__ pe) {
  int idx = blockIdx.x * blockDim.x + threadIdx.x;
  if (idx >= T_SEQ * DM) return;
  int t = idx >> 9, d = idx & (DM - 1);
  float f = __powf(1000.f, -(float)d / (float)DM);
  float a = (float)t * f;
  pe[idx] = (d & 1) ? __cosf(a) : __sinf(a);
}

__global__ void cvt_bf16_kernel(const float* __restrict__ src,
                                bf16* __restrict__ dst, int n) {
  int idx = blockIdx.x * blockDim.x + threadIdx.x;
  if (idx < n) dst[idx] = (bf16)src[idx];
}

__global__ void win_kernel(const float* __restrict__ pos, int* __restrict__ hw) {
  int idx = blockIdx.x * blockDim.x + threadIdx.x;
  if (idx >= B_SZ * T_SEQ) return;
  int b = idx >> 10, t = idx & (T_SEQ - 1);
  const float* pb = pos + (size_t)b * T_SEQ * 2;
  float vt = 0.f, vtm = 0.f, angt = 0.f, angtm = 0.f;
  if (t >= 1) {
    float dx = pb[2 * t] - pb[2 * (t - 1)];
    float dy = pb[2 * t + 1] - pb[2 * (t - 1) + 1];
    vt = sqrtf(dx * dx + dy * dy);
    angt = atan2f(dy, dx);
  }
  if (t >= 2) {
    float dx = pb[2 * (t - 1)] - pb[2 * (t - 2)];
    float dy = pb[2 * (t - 1) + 1] - pb[2 * (t - 2) + 1];
    vtm = sqrtf(dx * dx + dy * dy);
    angtm = atan2f(dy, dx);
  }
  float dv  = (t >= 1) ? fabsf(vt - vtm) : 0.f;
  float dth = (t >= 1) ? fabsf(angt - angtm) : 0.f;
  float Phi = 0.5f * dv + 0.5f * dth;
  float EF  = __expf(-(Phi - 0.6f) * (Phi - 0.6f) / (2.f * 0.3f * 0.3f));
  int w = (int)(102.f + EF);     // BS = int(0.1*1024) = 102
  hw[idx] = w >> 1;              // keep = (w >= 0.2) always true
}

// ---------------------------------------------------------------------------
// LayerNorm (ddof=1), wave-per-row, bf16 output
// ---------------------------------------------------------------------------
__global__ __launch_bounds__(256) void ln_kernel(const float* __restrict__ x,
                                                 const float* __restrict__ ga,
                                                 const float* __restrict__ gb,
                                                 bf16* __restrict__ y) {
  int row = blockIdx.x * 8 + (threadIdx.x >> 5);
  int l = threadIdx.x & 31;
  const float* xr = x + (size_t)row * DM;
  float4 v[4];
  float s = 0.f, ss = 0.f;
#pragma unroll
  for (int i = 0; i < 4; ++i) {
    v[i] = *(const float4*)&xr[i * 128 + l * 4];
    s  += v[i].x + v[i].y + v[i].z + v[i].w;
    ss += v[i].x * v[i].x + v[i].y * v[i].y + v[i].z * v[i].z + v[i].w * v[i].w;
  }
#pragma unroll
  for (int d = 1; d < 32; d <<= 1) {
    s += __shfl_xor(s, d, 32);
    ss += __shfl_xor(ss, d, 32);
  }
  float mean = s * (1.f / 512.f);
  float var = fmaxf(0.f, ss - 512.f * mean * mean) * (1.f / 511.f);
  float inv = 1.f / (sqrtf(var) + 1e-6f);
  bf16* yr = y + (size_t)row * DM;
#pragma unroll
  for (int i = 0; i < 4; ++i) {
    int n = i * 128 + l * 4;
    float4 g  = *(const float4*)&ga[n];
    float4 bb = *(const float4*)&gb[n];
    yr[n + 0] = (bf16)(g.x * (v[i].x - mean) * inv + bb.x);
    yr[n + 1] = (bf16)(g.y * (v[i].y - mean) * inv + bb.y);
    yr[n + 2] = (bf16)(g.z * (v[i].z - mean) * inv + bb.z);
    yr[n + 3] = (bf16)(g.w * (v[i].w - mean) * inv + bb.w);
  }
}

// ---------------------------------------------------------------------------
// bf16 WMMA GEMM:  C[M,N] = A[M,K] * W[N,K]^T + bias (+pe)(+resid)
// Block tile 128x64, 8 waves, each wave a 16x64 strip (4 WMMA C frags).
// Double-buffered LDS staging via async global->LDS DMA (ASYNCcnt).
// out: outF!=null -> f32 [M,N]; else bf16 scattered into [B,H,T,dk].
// ---------------------------------------------------------------------------
__global__ __launch_bounds__(256) void gemm_kernel(
    const bf16* __restrict__ A, const bf16* __restrict__ W,
    const float* __restrict__ bias, const float* __restrict__ resid,
    const float* __restrict__ pe, float* __restrict__ outF,
    bf16* __restrict__ outQKV, int M, int N, int K) {
  __shared__ bf16 As[2][128][40];
  __shared__ bf16 Ws[2][64][40];
  const int m0 = blockIdx.x * 128;
  const int n0 = blockIdx.y * 64;
  const int tid = threadIdx.x;
  const int w = tid >> 5, l = tid & 31;
  const int lrow = l & 15, lhi = l >> 4;
  const int wm = w * 16;

  f32x8 c0 = {}, c1 = {}, c2 = {}, c3 = {};
  const int ar0 = tid >> 2;
  const int ar1 = 64 + (tid >> 2);
  const int cc = (tid & 3) * 8;

  const int nk = K >> 5;
  // stage 0
  {
    async_copy16(&A[(size_t)(m0 + ar0) * K + cc], &As[0][ar0][cc]);
    async_copy16(&A[(size_t)(m0 + ar1) * K + cc], &As[0][ar1][cc]);
    async_copy16(&W[(size_t)(n0 + ar0) * K + cc], &Ws[0][ar0][cc]);
  }
  for (int kk = 0; kk < nk; ++kk) {
    const int cur = kk & 1;
    if (kk + 1 < nk) {
      const int k0 = (kk + 1) * 32;
      async_copy16(&A[(size_t)(m0 + ar0) * K + k0 + cc], &As[cur ^ 1][ar0][cc]);
      async_copy16(&A[(size_t)(m0 + ar1) * K + k0 + cc], &As[cur ^ 1][ar1][cc]);
      async_copy16(&W[(size_t)(n0 + ar0) * K + k0 + cc], &Ws[cur ^ 1][ar0][cc]);
      WAIT_ASYNC(3);   // in-order completion: stage kk is resident
    } else {
      WAIT_ASYNC(0);
    }
    __syncthreads();
    const bf16* ap = &As[cur][wm + lrow][lhi * 8];
    bf16x16 a = load_frag(ap, ap + 16);
    const bf16* b0p = &Ws[cur][ 0 + lrow][lhi * 8];
    const bf16* b1p = &Ws[cur][16 + lrow][lhi * 8];
    const bf16* b2p = &Ws[cur][32 + lrow][lhi * 8];
    const bf16* b3p = &Ws[cur][48 + lrow][lhi * 8];
    c0 = wmma_bf16(a, load_frag(b0p, b0p + 16), c0);
    c1 = wmma_bf16(a, load_frag(b1p, b1p + 16), c1);
    c2 = wmma_bf16(a, load_frag(b2p, b2p + 16), c2);
    c3 = wmma_bf16(a, load_frag(b3p, b3p + 16), c3);
    __syncthreads();
  }

  const f32x8 cf[4] = {c0, c1, c2, c3};
#pragma unroll
  for (int s = 0; s < 4; ++s) {
    int n = n0 + s * 16 + lrow;
#pragma unroll
    for (int j = 0; j < 8; ++j) {
      int m = m0 + wm + j + 8 * lhi;
      float val = cf[s][j] + bias[n];
      if (pe) val += pe[(size_t)(m & (T_SEQ - 1)) * DM + n];
      if (resid) val += resid[(size_t)m * N + n];
      if (outF) {
        outF[(size_t)m * N + n] = val;
      } else {
        int bb = m >> 10, tt = m & (T_SEQ - 1);
        outQKV[((((size_t)bb * NH) + (n >> 6)) * T_SEQ + tt) * DKH + (n & 63)] =
            (bf16)val;
      }
    }
  }
}

// ---------------------------------------------------------------------------
// Banded flash attention.  grid=(B*H, T/128), block=256 (8 waves, 16 q each).
// ---------------------------------------------------------------------------
__global__ __launch_bounds__(256) void attn_kernel(
    const bf16* __restrict__ Q, const bf16* __restrict__ Km,
    const bf16* __restrict__ Vm, const int* __restrict__ hw,
    bf16* __restrict__ attnout) {
  __shared__ bf16 Ks[32][72];       // keys x dk (async staged)
  __shared__ bf16 Vt[64][40];       // dk x keys (transposed via VGPR)
  __shared__ bf16 Pt[8][16][40];    // per-wave P transpose scratch
  const int bh = blockIdx.x;
  const int b = bh >> 3, h = bh & 7;
  const int qc0 = blockIdx.y * 128;
  const int tid = threadIdx.x;
  const int w = tid >> 5, l = tid & 31;
  const int lrow = l & 15, lhi = l >> 4;
  const int t0 = qc0 + w * 16;

  const bf16* Qp = Q + (size_t)bh * T_SEQ * DKH;
  const bf16* Kp = Km + (size_t)bh * T_SEQ * DKH;
  const bf16* Vp = Vm + (size_t)bh * T_SEQ * DKH;

  const bf16* qp = Qp + (size_t)(t0 + lrow) * DKH + lhi * 8;
  bf16x16 aq0 = load_frag(qp, qp + 16);
  bf16x16 aq1 = load_frag(qp + 32, qp + 48);

  float mrow[8], lsum[8];
  int hh[8];
  f32x8 a0 = {}, a1 = {}, a2 = {}, a3 = {};
#pragma unroll
  for (int j = 0; j < 8; ++j) {
    mrow[j] = -1e30f;
    lsum[j] = 0.f;
    hh[j] = hw[b * T_SEQ + t0 + j + 8 * lhi];
  }

  const int kstart = (qc0 > 52 ? (qc0 - 52) & ~31 : 0);
  const int kend = min(T_SEQ, qc0 + 128 + 52);   // half-width <= 52 by math

  for (int kb = kstart; kb < kend; kb += 32) {
    __syncthreads();
    {
      int key = tid >> 3, dc = (tid & 7) * 8;
      async_copy16(&Kp[(size_t)(kb + key) * DKH + dc], &Ks[key][dc]);
      bf16x8 vv = *(const bf16x8*)&Vp[(size_t)(kb + key) * DKH + dc];
#pragma unroll
      for (int i = 0; i < 8; ++i) Vt[dc + i][key] = vv[i];
    }
    WAIT_ASYNC(0);
    __syncthreads();

    f32x8 s0 = {}, s1 = {};
#pragma unroll
    for (int kk = 0; kk < 64; kk += 32) {
      bf16x16 a = (kk == 0) ? aq0 : aq1;
      const bf16* b0p = &Ks[lrow][kk + lhi * 8];
      const bf16* b1p = &Ks[16 + lrow][kk + lhi * 8];
      bf16x16 bb0 = load_frag(b0p, b0p + 16);
      bf16x16 bb1 = load_frag(b1p, b1p + 16);
      s0 = wmma_bf16(a, bb0, s0);
      s1 = wmma_bf16(a, bb1, s1);
    }

#pragma unroll
    for (int j = 0; j < 8; ++j) {
      int t = t0 + j + 8 * lhi;
      int k0i = kb + lrow, k1i = k0i + 16;
      bool v0 = (k0i >= t - hh[j]) && (k0i <= t + hh[j]);
      bool v1 = (k1i >= t - hh[j]) && (k1i <= t + hh[j]);
      float x0 = v0 ? s0[j] * 0.125f : -1e30f;
      float x1 = v1 ? s1[j] * 0.125f : -1e30f;
      float mx = fmaxf(x0, x1);
#pragma unroll
      for (int d = 1; d < 16; d <<= 1) mx = fmaxf(mx, __shfl_xor(mx, d, 32));
      float mnew = fmaxf(mrow[j], mx);
      float es = __expf(mrow[j] - mnew);
      float p0 = v0 ? __expf(x0 - mnew) : 0.f;
      float p1 = v1 ? __expf(x1 - mnew) : 0.f;
      float ps = p0 + p1;
#pragma unroll
      for (int d = 1; d < 16; d <<= 1) ps += __shfl_xor(ps, d, 32);
      lsum[j] = lsum[j] * es + ps;
      mrow[j] = mnew;
      a0[j] *= es; a1[j] *= es; a2[j] *= es; a3[j] *= es;
      int prow = j + 8 * lhi;
      Pt[w][prow][lrow] = (bf16)p0;
      Pt[w][prow][16 + lrow] = (bf16)p1;
    }

    const bf16* pp = &Pt[w][lrow][lhi * 8];
    bf16x16 ap = load_frag(pp, pp + 16);
    const bf16* v0p = &Vt[0 + lrow][lhi * 8];
    const bf16* v1p = &Vt[16 + lrow][lhi * 8];
    const bf16* v2p = &Vt[32 + lrow][lhi * 8];
    const bf16* v3p = &Vt[48 + lrow][lhi * 8];
    a0 = wmma_bf16(ap, load_frag(v0p, v0p + 16), a0);
    a1 = wmma_bf16(ap, load_frag(v1p, v1p + 16), a1);
    a2 = wmma_bf16(ap, load_frag(v2p, v2p + 16), a2);
    a3 = wmma_bf16(ap, load_frag(v3p, v3p + 16), a3);
  }

#pragma unroll
  for (int j = 0; j < 8; ++j) {
    float inv = 1.f / lsum[j];
    int t = t0 + j + 8 * lhi;
    size_t base = ((size_t)b * T_SEQ + t) * DM + h * DKH;
    attnout[base + 0 * 16 + lrow] = (bf16)(a0[j] * inv);
    attnout[base + 1 * 16 + lrow] = (bf16)(a1[j] * inv);
    attnout[base + 2 * 16 + lrow] = (bf16)(a2[j] * inv);
    attnout[base + 3 * 16 + lrow] = (bf16)(a3[j] * inv);
  }
}

// ---------------------------------------------------------------------------
// Host orchestration
// ---------------------------------------------------------------------------
static inline size_t align256(size_t x) { return (x + 255) & ~(size_t)255; }

extern "C" void kernel_launch(void* const* d_in, const int* in_sizes, int n_in,
                              void* d_out, int out_size, void* d_ws,
                              size_t ws_size, hipStream_t stream) {
  (void)in_sizes; (void)n_in; (void)out_size; (void)ws_size;
  const float* x   = (const float*)d_in[0];
  const float* pos = (const float*)d_in[1];
  const float* W0  = (const float*)d_in[2];
  const float* b0  = (const float*)d_in[3];
  auto LW  = [&](int i, int j) { return (const float*)d_in[4 + i * 14 + j * 2]; };
  auto LB  = [&](int i, int j) { return (const float*)d_in[4 + i * 14 + j * 2 + 1]; };
  auto LNA = [&](int i, int j) { return (const float*)d_in[4 + i * 14 + 10 + j * 2]; };
  auto LNB = [&](int i, int j) { return (const float*)d_in[4 + i * 14 + 11 + j * 2]; };

  char* p = (char*)d_ws;
  size_t off = 0;
  auto carve = [&](size_t bytes) { void* r = p + off; off = align256(off + bytes); return r; };
  float* pe     = (float*)carve((size_t)T_SEQ * DM * 4);
  int*   hw     = (int*)carve((size_t)B_SZ * T_SEQ * 4);
  bf16*  wbf    = (bf16*)carve((size_t)(32768 + 20 * 262144) * 2);
  float* hbuf   = (float*)carve((size_t)MROWS * DM * 4);
  bf16*  ybf    = (bf16*)carve((size_t)MROWS * DM * 2);  // also reused for x_bf16
  bf16*  Qbf    = (bf16*)carve((size_t)MROWS * DM * 2);
  bf16*  Kbf    = (bf16*)carve((size_t)MROWS * DM * 2);
  bf16*  Vbf    = (bf16*)carve((size_t)MROWS * DM * 2);
  bf16*  attnbf = (bf16*)carve((size_t)MROWS * DM * 2);

  pe_kernel<<<(T_SEQ * DM + 255) / 256, 256, 0, stream>>>(pe);
  win_kernel<<<(B_SZ * T_SEQ + 255) / 256, 256, 0, stream>>>(pos, hw);
  cvt_bf16_kernel<<<(32768 + 255) / 256, 256, 0, stream>>>(W0, wbf, 32768);
  for (int i = 0; i < 4; ++i)
    for (int j = 0; j < 5; ++j)
      cvt_bf16_kernel<<<(262144 + 255) / 256, 256, 0, stream>>>(
          LW(i, j), wbf + 32768 + (size_t)(i * 5 + j) * 262144, 262144);
  bf16* xbf = ybf;
  cvt_bf16_kernel<<<(MROWS * 64 + 255) / 256, 256, 0, stream>>>(x, xbf, MROWS * 64);

  gemm_kernel<<<dim3(MROWS / 128, DM / 64), 256, 0, stream>>>(
      xbf, wbf, b0, nullptr, pe, hbuf, nullptr, MROWS, DM, 64);

  for (int i = 0; i < 4; ++i) {
    bf16* wl = wbf + 32768 + (size_t)(i * 5) * 262144;
    ln_kernel<<<MROWS / 8, 256, 0, stream>>>(hbuf, LNA(i, 0), LNB(i, 0), ybf);
    gemm_kernel<<<dim3(MROWS / 128, DM / 64), 256, 0, stream>>>(
        ybf, wl + 0 * 262144, LB(i, 0), nullptr, nullptr, nullptr, Qbf, MROWS, DM, DM);
    gemm_kernel<<<dim3(MROWS / 128, DM / 64), 256, 0, stream>>>(
        ybf, wl + 1 * 262144, LB(i, 1), nullptr, nullptr, nullptr, Kbf, MROWS, DM, DM);
    gemm_kernel<<<dim3(MROWS / 128, DM / 64), 256, 0, stream>>>(
        ybf, wl + 2 * 262144, LB(i, 2), nullptr, nullptr, nullptr, Vbf, MROWS, DM, DM);
    attn_kernel<<<dim3(B_SZ * NH, T_SEQ / 128), 256, 0, stream>>>(
        Qbf, Kbf, Vbf, hw, attnbf);
    gemm_kernel<<<dim3(MROWS / 128, DM / 64), 256, 0, stream>>>(
        attnbf, wl + 3 * 262144, LB(i, 3), hbuf, nullptr, hbuf, nullptr, MROWS, DM, DM);
    ln_kernel<<<MROWS / 8, 256, 0, stream>>>(hbuf, LNA(i, 1), LNB(i, 1), ybf);
    float* dst = (i == 3) ? (float*)d_out : hbuf;
    gemm_kernel<<<dim3(MROWS / 128, DM / 64), 256, 0, stream>>>(
        ybf, wl + 4 * 262144, LB(i, 4), hbuf, nullptr, dst, nullptr, MROWS, DM, DM);
  }
}
